// DecoderBlock_12524124635966
// MI455X (gfx1250) — matrix operand order repaired and checked by reading, not compiled, not run on hardware
//
#include <hip/hip_runtime.h>

typedef __attribute__((ext_vector_type(16))) _Float16 v16h;
typedef __attribute__((ext_vector_type(8)))  _Float16 v8h;
typedef __attribute__((ext_vector_type(8)))  float    v8f;
typedef __attribute__((ext_vector_type(4)))  float    f4v;
typedef __attribute__((ext_vector_type(2)))  float    f2v;

#define ICH   64
#define IMH   128
#define IMW   128
#define KTOT  576          // 64 * 3 * 3
#define WPITCH 584         // weight LDS row pitch (halves): >=576, 16B-aligned, conflict-free stride
#define ICP    72          // tile ic pitch (halves): 144B lane stride -> conflict-free b128 loads
#define TX     66          // tile x extent: 64 pixels + 2 halo

// ---------------------------------------------------------------------------
// Kernel 1: 3x3 conv (64 -> 3) + bias + tanh via V_WMMA_F32_16X16X32_F16.
// Block = 128 threads = 4 waves, covers one (b,h) half-row of 64 pixels.
// K is logically permuted: k' = (ky*3+kx)*64 + ic, so B fragments are 16
// contiguous f16 in LDS (two ds_load_b128) and A fragments likewise.
// Inner loop: 4x ds_load_b128 + 1x wmma, no branches, no global traffic.
// ---------------------------------------------------------------------------
__global__ __launch_bounds__(128) void conv_tanh_wmma_kernel(
    const float* __restrict__ fused,   // [16,64,128,128]
    const float* __restrict__ conv_w,  // [3,64,3,3]
    const float* __restrict__ conv_b,  // [3]
    float* __restrict__ ll)            // [16,3,128,128]
{
    __shared__ __align__(16) _Float16 wlds[16 * WPITCH];    // A: [m][k'] (m>=3 zero)
    __shared__ __align__(16) _Float16 tile[3 * TX * ICP];   // B src: [ky][x][ic]

    const int tid = threadIdx.x;
    const int bid = blockIdx.x;                // 16*128*2 = 4096 blocks
    const int b   = bid >> 8;
    const int rem = bid & 255;
    const int h   = rem >> 1;
    const int wb  = (rem & 1) << 6;            // 0 or 64

    // ---- zero-fill weight LDS (rows 3..15 must be zero) ----
    for (int i = tid; i < 16 * WPITCH / 2; i += 128)
        ((unsigned int*)wlds)[i] = 0u;
    __syncthreads();

    // ---- scatter weights into permuted-K layout ----
    for (int i = tid; i < 3 * KTOT; i += 128) {
        const int m  = i / KTOT;
        const int r  = i - m * KTOT;           // ic*9 + ky*3 + kx
        const int ic = r / 9;
        const int r9 = r - ic * 9;             // ky*3 + kx
        wlds[m * WPITCH + r9 * 64 + ic] = (_Float16)conv_w[i];
    }

    // ---- stage input tile [ky][x][ic], x covers cols wb-1 .. wb+64, zero-padded ----
    for (int i = tid; i < 3 * TX * ICH; i += 128) {
        const int x  = i % TX;                 // fastest -> coalesced global reads
        const int t  = i / TX;
        const int ky = t % 3;
        const int ic = t / 3;
        const int iy = h + ky - 1;
        const int ix = wb + x - 1;
        float v = 0.0f;
        if (iy >= 0 && iy < IMH && ix >= 0 && ix < IMW)
            v = fused[(((size_t)b * ICH + ic) * IMH + iy) * IMW + ix];
        tile[(ky * TX + x) * ICP + ic] = (_Float16)v;
    }
    __syncthreads();

    // ---- WMMA main loop ----
    const int lane    = tid & 31;
    const int laneLow = lane & 15;
    const int laneHi  = lane >> 4;
    const int xw      = (tid >> 5) * 16 + laneLow;    // wave strip + pixel within strip

    const _Float16* wrow = wlds + laneLow * WPITCH;
    v8f acc = {};
    for (int g = 0; g < 9; ++g) {                     // g = ky*3 + kx
        const int ky = g / 3;
        const int kx = g - 3 * ky;
        const _Float16* bcol = tile + (ky * TX + xw + kx) * ICP;
#pragma unroll
        for (int half = 0; half < 2; ++half) {
            const int kc = g * 64 + half * 32;        // k' chunk base
            // A fragment: halves = k' {kc+laneHi*16 .. +15}
            const v8h a0 = *(const v8h*)(wrow + kc + laneHi * 16);
            const v8h a1 = *(const v8h*)(wrow + kc + laneHi * 16 + 8);
            // B fragment: halves = k' {kc+laneHi*16 .. +15} = 16 contiguous ic
            const int icb = half * 32 + laneHi * 16;
            const v8h b0 = *(const v8h*)(bcol + icb);
            const v8h b1 = *(const v8h*)(bcol + icb + 8);
            v16h a, bm;
#pragma unroll
            for (int i = 0; i < 8; ++i) {
                a[i] = a0[i];  a[i + 8] = a1[i];
                bm[i] = b0[i]; bm[i + 8] = b1[i];
            }
            acc = __builtin_amdgcn_wmma_f32_16x16x32_f16(
                false, a, false, bm, (short)0, acc, false, false);
        }
    }

    // C/D layout: VGPR m -> lanes 0-15 carry M=m, N=lane. Only m=0..2 real.
    if (laneHi == 0) {
        const float b0 = conv_b[0], b1 = conv_b[1], b2 = conv_b[2];
        const size_t base = ((size_t)b * 3) * (IMH * IMW) + (size_t)h * IMW + wb + xw;
        ll[base]                 = tanhf(acc[0] + b0);
        ll[base + IMH * IMW]     = tanhf(acc[1] + b1);
        ll[base + 2 * IMH * IMW] = tanhf(acc[2] + b2);
    }
}

// ---------------------------------------------------------------------------
// Kernel 2: both inverse Haar levels fused. One thread per coarse pixel
// (b,c,h,w) in [16,3,128,128] -> 4x4 output block in [16,3,512,512].
// Sign pattern: out = 0.5*(base ± lh ± hl ± hh); lh sign = row parity,
// hl sign = col parity, hh sign = (row==col). hf -> 2*hf - 1.
// hf1 reads and output writes are non-temporal (one-pass streams).
// ---------------------------------------------------------------------------
__global__ __launch_bounds__(256) void iwt2_fused_kernel(
    const float* __restrict__ ll,    // [16,3,128,128]
    const float* __restrict__ hf2,   // [16,9,128,128]
    const float* __restrict__ hf1,   // [16,9,256,256]
    float* __restrict__ out)         // [16,3,512,512]
{
    const int idx = blockIdx.x * 256 + threadIdx.x;   // 786432 total
    const int w = idx & 127;
    const int h = (idx >> 7) & 127;
    const int t = idx >> 14;          // b*3 + c
    const int c = t % 3;
    const int b = t / 3;

    const float llv = ll[idx];        // ll index == idx by construction

    const size_t h2 = (((size_t)b * 9 + 3 * c) << 14) + ((size_t)h << 7) + w;
    const float lh = 2.0f * hf2[h2]         - 1.0f;
    const float hl = 2.0f * hf2[h2 + 16384] - 1.0f;
    const float hh = 2.0f * hf2[h2 + 32768] - 1.0f;

    float curr[2][2];
#pragma unroll
    for (int p = 0; p < 2; ++p)
#pragma unroll
        for (int q = 0; q < 2; ++q)
            curr[p][q] = 0.5f * (llv + (p ? lh : -lh) + (q ? hl : -hl)
                                     + ((p == q) ? hh : -hh));

    const f2v* hf1v = (const f2v*)hf1;       // 8B-aligned pair loads
    float o[4][4];
#pragma unroll
    for (int p = 0; p < 2; ++p) {
        // float2 units: channel stride 256*256/2 = 32768
        const size_t rb = (((size_t)b * 9 + 3 * c) * 256 + (2 * h + p)) * 128 + w;
        const f2v l2 = __builtin_nontemporal_load(hf1v + rb);
        const f2v g2 = __builtin_nontemporal_load(hf1v + rb + 32768);
        const f2v q2 = __builtin_nontemporal_load(hf1v + rb + 65536);
        const float L[2] = {2.0f * l2.x - 1.0f, 2.0f * l2.y - 1.0f};
        const float G[2] = {2.0f * g2.x - 1.0f, 2.0f * g2.y - 1.0f};
        const float Q[2] = {2.0f * q2.x - 1.0f, 2.0f * q2.y - 1.0f};
#pragma unroll
        for (int q = 0; q < 2; ++q) {
            const float cur = curr[p][q];
#pragma unroll
            for (int r = 0; r < 2; ++r)
#pragma unroll
                for (int s = 0; s < 2; ++s)
                    o[2 * p + r][2 * q + s] =
                        0.5f * (cur + (r ? L[q] : -L[q]) + (s ? G[q] : -G[q])
                                    + ((r == s) ? Q[q] : -Q[q]));
        }
    }

    // 4 rows of 4 consecutive floats -> b128 NT stores, contiguous across a wave
    const size_t ob = (((size_t)b * 3 + c) * 512 + 4 * (size_t)h) * 512 + 4 * w;
#pragma unroll
    for (int row = 0; row < 4; ++row) {
        f4v v4 = {o[row][0], o[row][1], o[row][2], o[row][3]};
        __builtin_nontemporal_store(v4, (f4v*)(out + ob + (size_t)row * 512));
    }
}

extern "C" void kernel_launch(void* const* d_in, const int* in_sizes, int n_in,
                              void* d_out, int out_size, void* d_ws, size_t ws_size,
                              hipStream_t stream) {
    const float* fused  = (const float*)d_in[0];
    const float* hf1    = (const float*)d_in[1];
    const float* hf2    = (const float*)d_in[2];
    const float* conv_w = (const float*)d_in[3];
    const float* conv_b = (const float*)d_in[4];
    float* out = (float*)d_out;
    float* ll  = (float*)d_ws;   // 16*3*128*128 floats = 3 MiB scratch

    (void)in_sizes; (void)n_in; (void)out_size; (void)ws_size;

    // 4096 blocks x 128 threads: one (b,h) half-row (64 px, 4 wave-strips) each
    conv_tanh_wmma_kernel<<<4096, 128, 0, stream>>>(fused, conv_w, conv_b, ll);
    // 786432 coarse pixels, one thread each
    iwt2_fused_kernel<<<3072, 256, 0, stream>>>(ll, hf2, hf1, out);
}
